// SelfAttention_31671088841431
// MI455X (gfx1250) — matrix operand order repaired and checked by reading, not compiled, not run on hardware
//
#include <hip/hip_runtime.h>

// ---------------- problem constants ----------------
#define EMB   1024
#define NHD   16
#define HSZ   64
#define TSEQ  2048
#define BATCH 4
#define MTOT  (BATCH * TSEQ)   // 8192
#define NQKV  (3 * EMB)        // 3072

typedef __attribute__((ext_vector_type(16))) __bf16 v16bf;
typedef __attribute__((ext_vector_type(8)))  float  v8f;
typedef __attribute__((ext_vector_type(8)))  __bf16 v8bf;
typedef __attribute__((ext_vector_type(4)))  unsigned v4u;
typedef __attribute__((ext_vector_type(4)))  int    v4i;
typedef __attribute__((ext_vector_type(8)))  int    v8i;

static __device__ __forceinline__ __bf16 f2bf(float f) {
    union { float f; unsigned u; } in; in.f = f;
    unsigned r = in.u + 0x7FFFu + ((in.u >> 16) & 1u);   // round-to-nearest-even
    union { unsigned short s; __bf16 b; } out;
    out.s = (unsigned short)(r >> 16);
    return out.b;
}

// ---------------------------------------------------------------------------
// 16-lane (DPP row) reductions, pure VALU: row_ror 8/4/2/1 rotate-reduce.
// Each 16-lane half of the wave reduces independently -- matches the lane
// group that owns one row set of a WMMA C fragment.
// ---------------------------------------------------------------------------
static __device__ __forceinline__ float row16_max(float v) {
    union { float f; int i; } a, b;
    a.f = v;
    b.i = __builtin_amdgcn_update_dpp(0, a.i, 0x128, 0xF, 0xF, true); a.f = fmaxf(a.f, b.f); // ror:8
    b.i = __builtin_amdgcn_update_dpp(0, a.i, 0x124, 0xF, 0xF, true); a.f = fmaxf(a.f, b.f); // ror:4
    b.i = __builtin_amdgcn_update_dpp(0, a.i, 0x122, 0xF, 0xF, true); a.f = fmaxf(a.f, b.f); // ror:2
    b.i = __builtin_amdgcn_update_dpp(0, a.i, 0x121, 0xF, 0xF, true); a.f = fmaxf(a.f, b.f); // ror:1
    return a.f;
}
static __device__ __forceinline__ float row16_sum(float v) {
    union { float f; int i; } a, b;
    a.f = v;
    b.i = __builtin_amdgcn_update_dpp(0, a.i, 0x128, 0xF, 0xF, true); a.f += b.f;
    b.i = __builtin_amdgcn_update_dpp(0, a.i, 0x124, 0xF, 0xF, true); a.f += b.f;
    b.i = __builtin_amdgcn_update_dpp(0, a.i, 0x122, 0xF, 0xF, true); a.f += b.f;
    b.i = __builtin_amdgcn_update_dpp(0, a.i, 0x121, 0xF, 0xF, true); a.f += b.f;
    return a.f;
}

// ---------------------------------------------------------------------------
// TDM: issue one 2D bf16 tile load Global->LDS via Tensor Data Mover.
// ---------------------------------------------------------------------------
static __device__ __forceinline__ void tdm_load_tile_bf16(unsigned lds_off,
                                                          const __bf16* gptr,
                                                          unsigned tile_k,     // dim0 (contiguous elems)
                                                          unsigned tile_rows,  // dim1
                                                          unsigned row_stride) // elems
{
    const unsigned long long ga = (unsigned long long)(size_t)gptr;
    v4u g0;
    g0.x = 1u;                                              // count=1, user mode
    g0.y = lds_off;                                         // LDS byte address
    g0.z = (unsigned)(ga & 0xFFFFFFFFull);                  // global_addr[31:0]
    g0.w = (unsigned)((ga >> 32) & 0x01FFFFFFull) | (2u << 30); // addr[56:32] | type=2
    const unsigned td0 = row_stride;                        // tensor_dim0 = full row
    const unsigned td1 = 0x00100000u;                       // tensor_dim1: large (no OOB)
    v8i g1;
    g1[0] = (int)(1u << 16);                                // data_size=1 (2 bytes)
    g1[1] = (int)((td0 & 0xFFFFu) << 16);                   // tensor_dim0[15:0]
    g1[2] = (int)((td0 >> 16) | ((td1 & 0xFFFFu) << 16));   // dim0[31:16] | dim1[15:0]
    g1[3] = (int)((td1 >> 16) | (tile_k << 16));            // dim1[31:16] | tile_dim0
    g1[4] = (int)(tile_rows & 0xFFFFu);                     // tile_dim1 (tile_dim2=0)
    g1[5] = (int)row_stride;                                // tensor_dim0_stride[31:0]
    g1[6] = 0;
    g1[7] = 0;
    const v4i z4 = {0, 0, 0, 0};
#if defined(__clang_major__) && (__clang_major__ >= 23)
    const v8i z8 = {0, 0, 0, 0, 0, 0, 0, 0};
    __builtin_amdgcn_tensor_load_to_lds(g0, g1, z4, z4, z8, 0);
#else
    __builtin_amdgcn_tensor_load_to_lds(g0, g1, z4, z4, 0);
#endif
}

static __device__ __forceinline__ unsigned lds_offset_of(const void* p) {
    return (unsigned)(unsigned long long)(size_t)p;   // generic LDS addr low 32 bits = offset
}

// ---------------------------------------------------------------------------
// Elementwise f32 -> bf16 (for x). 8 elems/thread.
// ---------------------------------------------------------------------------
__global__ __launch_bounds__(256)
void cvt_bf16(const float* __restrict__ in, __bf16* __restrict__ out)
{
    const int i = (blockIdx.x * 256 + threadIdx.x) * 8;
    const float4 a = *(const float4*)(in + i);
    const float4 b = *(const float4*)(in + i + 4);
    v8bf o;
    o[0] = f2bf(a.x); o[1] = f2bf(a.y); o[2] = f2bf(a.z); o[3] = f2bf(a.w);
    o[4] = f2bf(b.x); o[5] = f2bf(b.y); o[6] = f2bf(b.z); o[7] = f2bf(b.w);
    *(v8bf*)(out + i) = o;
}

// ---------------------------------------------------------------------------
// Weight transpose + convert: w[K][N] f32 -> wT[N][K] bf16 (32x32 LDS tiles).
// ---------------------------------------------------------------------------
__global__ __launch_bounds__(256)
void transpose_bf16(const float* __restrict__ w, __bf16* __restrict__ wt, int K, int N)
{
    __shared__ __bf16 tile[32][33];
    const int tx = threadIdx.x & 31;
    const int ty = threadIdx.x >> 5;            // 0..7
    const int n0 = blockIdx.x * 32;
    const int k0 = blockIdx.y * 32;
    #pragma unroll
    for (int j = 0; j < 4; ++j) {
        const int kk = ty + j * 8;
        tile[kk][tx] = f2bf(w[(size_t)(k0 + kk) * N + n0 + tx]);
    }
    __syncthreads();
    #pragma unroll
    for (int j = 0; j < 4; ++j) {
        const int nn = ty + j * 8;
        wt[(size_t)(n0 + nn) * K + k0 + tx] = tile[tx][nn];
    }
}

static __device__ __forceinline__ void qkv_scatter(__bf16* __restrict__ q,
                                                   __bf16* __restrict__ k,
                                                   __bf16* __restrict__ vt,
                                                   int m, int n, float val) {
    const int b   = m >> 11;
    const int t   = m & (TSEQ - 1);
    const int seg = n >> 10;                   // 0=q 1=k 2=v
    const int c   = n & (EMB - 1);
    const int h   = c >> 6;
    const int d   = c & (HSZ - 1);
    const int bh  = b * NHD + h;
    if (seg == 0)      q [((size_t)bh * TSEQ + t) * HSZ + d] = f2bf(val * 0.125f); // 1/sqrt(64)
    else if (seg == 1) k [((size_t)bh * TSEQ + t) * HSZ + d] = f2bf(val);
    else               vt[((size_t)bh * HSZ  + d) * TSEQ + t] = f2bf(val);         // V transposed
}

// ---------------------------------------------------------------------------
// GEMM: C[M,N] = A[M,K](bf16,row) * Bt[N,K](bf16,row)^T + bias.
// Block 256 thr = 8 waves; tile 128x64, BK=32; wave = 32x32 (4 WMMA/k-step).
// Tiles staged by TDM (wave 0 issues descriptors), double-buffered LDS.
// ---------------------------------------------------------------------------
template <int MODE>
__global__ __launch_bounds__(256)
void gemm_bf16(const __bf16* __restrict__ A, const __bf16* __restrict__ Bt,
               const float* __restrict__ bias, float* __restrict__ outF,
               __bf16* __restrict__ qO, __bf16* __restrict__ kO,
               __bf16* __restrict__ vtO, int M, int N, int K)
{
    __shared__ __bf16 ldsA[2][128 * 32];   // [row][k]   16 KB
    __shared__ __bf16 ldsB[2][64 * 32];    // [n][k]      8 KB

    const int tid   = threadIdx.x;
    const int lane  = tid & 31;
    const int wid   = tid >> 5;
    const int ln    = lane & 15;
    const int hi    = lane >> 4;
    const int waveM = wid >> 1;            // 0..3 -> 32-row strip
    const int waveN = wid & 1;             // 0..1 -> 32-col strip
    const int m0    = blockIdx.x * 128;
    const int n0    = blockIdx.y * 64;
    const int ksteps = K >> 5;

    v8f c00 = {0.f,0.f,0.f,0.f,0.f,0.f,0.f,0.f};
    v8f c01 = c00, c10 = c00, c11 = c00;

    if (wid == 0) {
        tdm_load_tile_bf16(lds_offset_of(&ldsA[0][0]), A  + (size_t)m0 * K, 32, 128, (unsigned)K);
        tdm_load_tile_bf16(lds_offset_of(&ldsB[0][0]), Bt + (size_t)n0 * K, 32,  64, (unsigned)K);
    }

    for (int i = 0; i < ksteps; ++i) {
        const int st = i & 1;
        if (wid == 0) {
            if (i + 1 < ksteps) {
                const int k1 = (i + 1) << 5;
                tdm_load_tile_bf16(lds_offset_of(&ldsA[st ^ 1][0]),
                                   A  + (size_t)m0 * K + k1, 32, 128, (unsigned)K);
                tdm_load_tile_bf16(lds_offset_of(&ldsB[st ^ 1][0]),
                                   Bt + (size_t)n0 * K + k1, 32,  64, (unsigned)K);
                __builtin_amdgcn_s_wait_tensorcnt((short)2);   // stage i landed
            } else {
                __builtin_amdgcn_s_wait_tensorcnt((short)0);
            }
        }
        __syncthreads();

        const __bf16* lA = &ldsA[st][0];
        const __bf16* lB = &ldsB[st][0];
        v16bf af0, af1, b0, b1;
        #pragma unroll
        for (int g = 0; g < 8; ++g) {
            const int kb = 2 * g + hi * 16;   // A-frag K map
            af0[2*g]   = lA[(waveM * 32 + ln) * 32 + kb];
            af0[2*g+1] = lA[(waveM * 32 + ln) * 32 + kb + 1];
            af1[2*g]   = lA[(waveM * 32 + 16 + ln) * 32 + kb];
            af1[2*g+1] = lA[(waveM * 32 + 16 + ln) * 32 + kb + 1];
            const int kk = 2 * g + hi * 16;                           // B-frag K map
            b0[2*g]   = lB[(waveN * 32 + ln) * 32 + kk];
            b0[2*g+1] = lB[(waveN * 32 + ln) * 32 + kk + 1];
            b1[2*g]   = lB[(waveN * 32 + 16 + ln) * 32 + kk];
            b1[2*g+1] = lB[(waveN * 32 + 16 + ln) * 32 + kk + 1];
        }
        c00 = __builtin_amdgcn_wmma_f32_16x16x32_bf16(false, af0, false, b0, (short)0, c00, false, false);
        c01 = __builtin_amdgcn_wmma_f32_16x16x32_bf16(false, af0, false, b1, (short)0, c01, false, false);
        c10 = __builtin_amdgcn_wmma_f32_16x16x32_bf16(false, af1, false, b0, (short)0, c10, false, false);
        c11 = __builtin_amdgcn_wmma_f32_16x16x32_bf16(false, af1, false, b1, (short)0, c11, false, false);
        __syncthreads();
    }

    #pragma unroll
    for (int mf = 0; mf < 2; ++mf) {
        const v8f& cl = mf ? c10 : c00;
        const v8f& ch = mf ? c11 : c01;
        #pragma unroll
        for (int r = 0; r < 8; ++r) {
            const int row  = m0 + waveM * 32 + mf * 16 + r + 8 * hi;
            const int col0 = n0 + waveN * 32 + ln;
            const int col1 = col0 + 16;
            const float v0 = cl[r] + bias[col0];
            const float v1 = ch[r] + bias[col1];
            if (MODE == 0) {
                outF[(size_t)row * N + col0] = v0;
                outF[(size_t)row * N + col1] = v1;
            } else {
                qkv_scatter(qO, kO, vtO, row, col0, v0);
                qkv_scatter(qO, kO, vtO, row, col1, v1);
            }
        }
    }
}

// ---------------------------------------------------------------------------
// Flash attention: one wave per (b, h, 16-query tile). bf16 output [B,T,C].
// ---------------------------------------------------------------------------
__global__ __launch_bounds__(32)
void attn_fwd(const __bf16* __restrict__ q, const __bf16* __restrict__ k,
              const __bf16* __restrict__ vt, __bf16* __restrict__ attn)
{
    __shared__ __bf16 ldsP[16 * 32];

    const int lane = threadIdx.x;
    const int ln   = lane & 15;
    const int hi   = lane >> 4;
    const int qt   = blockIdx.x & (TSEQ / 16 - 1);
    const int bh   = blockIdx.x >> 7;
    const int b    = bh >> 4;
    const int h    = bh & 15;
    const int q0   = qt * 16;

    const __bf16* qb = q  + (size_t)bh * TSEQ * HSZ;
    const __bf16* kb = k  + (size_t)bh * TSEQ * HSZ;
    const __bf16* vb = vt + (size_t)bh * HSZ  * TSEQ;

    v16bf qa0, qa1;
    #pragma unroll
    for (int g = 0; g < 8; ++g) {
        const int kbv = 2 * g + hi * 16;
        qa0[2*g]   = qb[(size_t)(q0 + ln) * HSZ + kbv];
        qa0[2*g+1] = qb[(size_t)(q0 + ln) * HSZ + kbv + 1];
        qa1[2*g]   = qb[(size_t)(q0 + ln) * HSZ + 32 + kbv];
        qa1[2*g+1] = qb[(size_t)(q0 + ln) * HSZ + 32 + kbv + 1];
    }

    float mrow[8], lrow[8];
    v8f o0 = {0.f,0.f,0.f,0.f,0.f,0.f,0.f,0.f};
    v8f o1 = o0, o2 = o0, o3 = o0;
    #pragma unroll
    for (int r = 0; r < 8; ++r) { mrow[r] = -3.0e38f; lrow[r] = 0.f; }

    for (int c = 0; c < TSEQ / 32; ++c) {
        const int key0 = c * 32;
        if (c + 1 < TSEQ / 32) {                 // pull next chunk toward L2/WGP$
            __builtin_prefetch(kb + (size_t)(key0 + 32 + lane) * HSZ, 0, 1);
            __builtin_prefetch(vb + (size_t)(lane * 2) * TSEQ + key0 + 32, 0, 1);
        }
        v8f s0 = {0.f,0.f,0.f,0.f,0.f,0.f,0.f,0.f};
        v8f s1 = s0;

        #pragma unroll
        for (int ks = 0; ks < 2; ++ks) {
            v16bf bk0, bk1;
            #pragma unroll
            for (int g = 0; g < 8; ++g) {
                const int kk = ks * 32 + 2 * g + hi * 16;
                bk0[2*g]   = kb[(size_t)(key0 + ln) * HSZ + kk];
                bk0[2*g+1] = kb[(size_t)(key0 + ln) * HSZ + kk + 1];
                bk1[2*g]   = kb[(size_t)(key0 + 16 + ln) * HSZ + kk];
                bk1[2*g+1] = kb[(size_t)(key0 + 16 + ln) * HSZ + kk + 1];
            }
            const v16bf& qa = (ks == 0) ? qa0 : qa1;
            s0 = __builtin_amdgcn_wmma_f32_16x16x32_bf16(false, qa, false, bk0, (short)0, s0, false, false);
            s1 = __builtin_amdgcn_wmma_f32_16x16x32_bf16(false, qa, false, bk1, (short)0, s1, false, false);
        }

        // ---- online softmax; row reductions via DPP rotate-reduce ----
        float alpha[8];
        #pragma unroll
        for (int r = 0; r < 8; ++r) {
            const float mc = row16_max(fmaxf(s0[r], s1[r]));
            const float mn = fmaxf(mrow[r], mc);
            alpha[r] = __expf(mrow[r] - mn);
            const float p0 = __expf(s0[r] - mn);
            const float p1 = __expf(s1[r] - mn);
            s0[r] = p0; s1[r] = p1;
            const float rs = row16_sum(p0 + p1);
            lrow[r] = lrow[r] * alpha[r] + rs;
            mrow[r] = mn;
        }

        #pragma unroll
        for (int r = 0; r < 8; ++r) {
            ldsP[(r + 8 * hi) * 32 + ln]      = f2bf(s0[r]);
            ldsP[(r + 8 * hi) * 32 + 16 + ln] = f2bf(s1[r]);
        }
        __syncthreads();
        v16bf pa;
        #pragma unroll
        for (int g = 0; g < 8; ++g) {
            const int kbv = 2 * g + hi * 16;
            pa[2*g]   = ldsP[ln * 32 + kbv];
            pa[2*g+1] = ldsP[ln * 32 + kbv + 1];
        }
        __syncthreads();

        #pragma unroll
        for (int r = 0; r < 8; ++r) {
            o0[r] *= alpha[r]; o1[r] *= alpha[r];
            o2[r] *= alpha[r]; o3[r] *= alpha[r];
        }
        #pragma unroll
        for (int f = 0; f < 4; ++f) {
            v16bf bv;
            #pragma unroll
            for (int g = 0; g < 8; ++g) {
                const int kk = 2 * g + hi * 16;
                const int n  = f * 16 + ln;
                bv[2*g]   = vb[(size_t)n * TSEQ + key0 + kk];
                bv[2*g+1] = vb[(size_t)n * TSEQ + key0 + kk + 1];
            }
            if      (f == 0) o0 = __builtin_amdgcn_wmma_f32_16x16x32_bf16(false, pa, false, bv, (short)0, o0, false, false);
            else if (f == 1) o1 = __builtin_amdgcn_wmma_f32_16x16x32_bf16(false, pa, false, bv, (short)0, o1, false, false);
            else if (f == 2) o2 = __builtin_amdgcn_wmma_f32_16x16x32_bf16(false, pa, false, bv, (short)0, o2, false, false);
            else             o3 = __builtin_amdgcn_wmma_f32_16x16x32_bf16(false, pa, false, bv, (short)0, o3, false, false);
        }
    }

    #pragma unroll
    for (int r = 0; r < 8; ++r) {
        const float inv = 1.0f / lrow[r];
        const int t = q0 + r + 8 * hi;
        const size_t base = ((size_t)(b * TSEQ + t)) * EMB + h * HSZ;
        attn[base +  0 + ln] = f2bf(o0[r] * inv);
        attn[base + 16 + ln] = f2bf(o1[r] * inv);
        attn[base + 32 + ln] = f2bf(o2[r] * inv);
        attn[base + 48 + ln] = f2bf(o3[r] * inv);
    }
}

// ---------------------------------------------------------------------------
extern "C" void kernel_launch(void* const* d_in, const int* in_sizes, int n_in,
                              void* d_out, int out_size, void* d_ws, size_t ws_size,
                              hipStream_t stream)
{
    const float* x     = (const float*)d_in[0];
    const float* w_qkv = (const float*)d_in[1];
    const float* b_qkv = (const float*)d_in[2];
    const float* w_out = (const float*)d_in[3];
    const float* b_out = (const float*)d_in[4];
    float* out = (float*)d_out;

    char* ws = (char*)d_ws;
    const size_t MB = 1024 * 1024;
    __bf16* qbuf  = (__bf16*)(ws);               // 16 MB  [B,NH,T,HS] (scaled)
    __bf16* kbuf  = (__bf16*)(ws + 16 * MB);     // 16 MB  [B,NH,T,HS]
    __bf16* vtbuf = (__bf16*)(ws + 32 * MB);     // 16 MB  [B,NH,HS,T]
    __bf16* attnb = (__bf16*)(ws + 48 * MB);     // 16 MB  [B,T,C]
    __bf16* xb    = (__bf16*)(ws + 64 * MB);     // 16 MB  x in bf16
    __bf16* wqkvT = (__bf16*)(ws + 80 * MB);     //  6 MB  [3072][1024]
    __bf16* woutT = (__bf16*)(ws + 88 * MB);     //  2 MB  [1024][1024]

    // 0) precision prep: x -> bf16; weights -> transposed bf16 [N][K]
    cvt_bf16<<<dim3((MTOT * EMB) / (256 * 8)), dim3(256), 0, stream>>>(x, xb);
    transpose_bf16<<<dim3(NQKV / 32, EMB / 32), dim3(256), 0, stream>>>(w_qkv, wqkvT, EMB, NQKV);
    transpose_bf16<<<dim3(EMB / 32,  EMB / 32), dim3(256), 0, stream>>>(w_out, woutT, EMB, EMB);

    // 1) QKV projection (TDM-staged WMMA GEMM) + bias + head split
    gemm_bf16<1><<<dim3(MTOT / 128, NQKV / 64), dim3(256), 0, stream>>>(
        xb, wqkvT, b_qkv, nullptr, qbuf, kbuf, vtbuf, MTOT, NQKV, EMB);

    // 2) flash attention
    attn_fwd<<<dim3(BATCH * NHD * (TSEQ / 16)), dim3(32), 0, stream>>>(qbuf, kbuf, vtbuf, attnb);

    // 3) output projection + bias -> f32 d_out
    gemm_bf16<0><<<dim3(MTOT / 128, EMB / 64), dim3(256), 0, stream>>>(
        attnb, woutT, b_out, out, nullptr, nullptr, nullptr, MTOT, EMB, EMB);
}
